// KVAttentionLayer_11553462026817
// MI455X (gfx1250) — matrix operand order repaired and checked by reading, not compiled, not run on hardware
//
#include <hip/hip_runtime.h>

// Problem constants (match reference: B=8, N=2048, D=128, edge types 1..4)
constexpr int Bc = 8;
constexpr int Nc = 2048;
constexpr int Dc = 128;
constexpr float LEAKY = 0.2f;
constexpr float MASKF = -9e15f;

typedef __attribute__((ext_vector_type(16))) __bf16 v16bf;
typedef __attribute__((ext_vector_type(8)))  float  v8f;

union FragBF {          // one 16x32 bf16 WMMA operand fragment: 8 VGPRs/lane
    v16bf v;
    uint4 u[2];
    uint2 w[4];
};

__device__ __forceinline__ unsigned short f2bf(float f) {
    unsigned u = __float_as_uint(f);
    u += 0x7fffu + ((u >> 16) & 1u);     // round-to-nearest-even
    return (unsigned short)(u >> 16);
}

// 16-lane xor-butterfly steps with DPP fused directly into the ALU op
// (one v_max_num_f32_dpp / v_add_f32_dpp per step instead of mov+op).
#define DPP_MAXX(var, MASK)                                                   \
    asm("v_max_num_f32_dpp %0, %1, %1 row_xmask:" #MASK                       \
        " row_mask:0xf bank_mask:0xf"                                         \
        : "=v"(var) : "v"(var))
#define DPP_ADDX(var, MASK)                                                   \
    asm("v_add_f32_dpp %0, %1, %1 row_xmask:" #MASK                           \
        " row_mask:0xf bank_mask:0xf"                                         \
        : "=v"(var) : "v"(var))

// ---------------------------------------------------------------------------
// Kernel 1: per-node-type projection q[b,n,:] = hidden[b,n,:] @ W_Q[type]
// (type 0 -> zero row, matching the reference one_hot(-1) behavior).
// ---------------------------------------------------------------------------
__global__ __launch_bounds__(Dc) void proj_kernel(
    const float* __restrict__ hidden,
    const int*   __restrict__ ntype,
    const float* __restrict__ WQ,
    unsigned short* __restrict__ qbf)
{
    const int row = blockIdx.x;          // 0 .. B*N-1
    const int j   = threadIdx.x;         // output column 0..127
    __shared__ float h[Dc];
    h[j] = hidden[(size_t)row * Dc + j];
    __syncthreads();

    const int t = ntype[row];
    float acc = 0.0f;
    if (t != 0) {
        const float* w = WQ + (size_t)t * Dc * Dc + j;   // W[t, k, j]
        #pragma unroll 8
        for (int k = 0; k < Dc; ++k)
            acc = fmaf(h[k], w[(size_t)k * Dc], acc);
    }
    qbf[(size_t)row * Dc + j] = f2bf(acc);
}

// ---------------------------------------------------------------------------
// Kernel 2: fused masked-attention.  Block = 128 threads = 4 waves.
// Wave w owns 16 query rows; block loops over 64 key-tile pairs (32 columns
// per iteration) with double-buffered LDS staging of the 32x128 q_J tile
// (row-major copy for score B-fragments + transposed copy for P.V).
// Softmax denominator comes free from a ones-column WMMA accumulator.
// ---------------------------------------------------------------------------
constexpr int QJ_STRIDE = 136;  // bf16 elems; 272B rows, 16B aligned, conflict-free
constexpr int VT_STRIDE = 36;   // bf16 elems; 72B rows, 8B aligned
constexpr int PT_STRIDE = 40;   // bf16 elems; 80B rows, 16B aligned

__global__ __launch_bounds__(128) void attn_kernel(
    const unsigned short* __restrict__ qbf,
    const int*            __restrict__ adj,
    float*                __restrict__ out)
{
    __shared__ unsigned short qj[2][32 * QJ_STRIDE];     // q_J pair, row-major
    __shared__ unsigned short vt[2][Dc * VT_STRIDE];     // q_J pair, transposed
    __shared__ unsigned short pt[4 * 16 * PT_STRIDE];    // per-wave P (16x32)

    const int tid  = threadIdx.x;
    const int wave = tid >> 5;
    const int lane = tid & 31;
    const int hh   = lane >> 4;      // half: 0 or 1
    const int ln   = lane & 15;

    const int b  = blockIdx.x >> 5;              // 32 blocks per batch (N/64)
    const int I0 = (blockIdx.x & 31) * 64 + wave * 16;

    const unsigned short* qb = qbf + (size_t)b * Nc * Dc;

    // Resident A fragments of Q_I (16 rows x 128 K, 4 ksteps of 32)
    FragBF aQ[4];
    {
        const unsigned short* rowp = qb + (size_t)(I0 + ln) * Dc;
        #pragma unroll
        for (int ks = 0; ks < 4; ++ks) {
            aQ[ks].u[0] = *(const uint4*)(rowp + ks * 32 + 8 * hh);       // K 8h..8h+7
            aQ[ks].u[1] = *(const uint4*)(rowp + ks * 32 + 16 + 8 * hh);  // K 16+8h..
        }
    }

    // Constant ones-column B fragment: column 0 = 1.0 for every K, rest 0.
    FragBF ones;
    {
        const unsigned po = (ln == 0) ? 0x3F803F80u : 0u;   // two bf16 1.0
        ones.u[0] = make_uint4(po, po, po, po);
        ones.u[1] = make_uint4(po, po, po, po);
    }

    v8f acc[8], acc9;                 // output tiles + row-sum accumulator
    #pragma unroll
    for (int nt = 0; nt < 8; ++nt)
        #pragma unroll
        for (int e = 0; e < 8; ++e) acc[nt][e] = 0.0f;
    #pragma unroll
    for (int e = 0; e < 8; ++e) acc9[e] = 0.0f;

    float mrun[8];
    int   adjbase[8];
    #pragma unroll
    for (int r = 0; r < 8; ++r) {
        mrun[r] = -3.0e38f;
        adjbase[r] = (b * Nc + I0 + r + 8 * hh) * Nc;   // fits in int (33.5M)
    }

    unsigned short* ptw = &pt[wave * 16 * PT_STRIDE];

    // Cooperative staging decomposition: 128 threads, 32 rows x 128 cols
    const int jp  = (tid >> 4) & 7;   // row-pair 0..7 within a 16-row half
    const int seg = tid & 15;         // 8-column segment 0..15

    // ---- prologue: stage buffer 0 for Jp = 0 -----------------------------
    #pragma unroll
    for (int t2 = 0; t2 < 2; ++t2) {
        const int jr = t2 * 16 + 2 * jp;
        const unsigned short* gp = qb + (size_t)jr * Dc + seg * 8;
        uint4 x0 = *(const uint4*)(gp);
        uint4 x1 = *(const uint4*)(gp + Dc);
        *(uint4*)&qj[0][jr * QJ_STRIDE + seg * 8]       = x0;
        *(uint4*)&qj[0][(jr + 1) * QJ_STRIDE + seg * 8] = x1;
        const unsigned short* a0 = (const unsigned short*)&x0;
        const unsigned short* a1 = (const unsigned short*)&x1;
        unsigned* vt32 = (unsigned*)&vt[0][0];
        #pragma unroll
        for (int e = 0; e < 8; ++e) {
            const int d = seg * 8 + e;
            vt32[d * (VT_STRIDE / 2) + 8 * t2 + jp] =
                (unsigned)a0[e] | ((unsigned)a1[e] << 16);
        }
    }

    constexpr int NT = Nc / 32;       // 64 iterations
    #pragma unroll 1
    for (int Jp = 0; Jp < NT; ++Jp) {
        const int Jc  = Jp * 32;
        const int cur = Jp & 1;
        const unsigned short* qjc = &qj[cur][0];
        const unsigned short* vtc = &vt[cur][0];

        __syncthreads();   // staging of cur visible; nxt buffer reads retired

        // ---- issue global loads for tile Jp+1 (latency hidden by compute)
        uint4 n00 = {}, n01 = {}, n10 = {}, n11 = {};
        const bool hasNext = (Jp + 1 < NT);
        if (hasNext) {
            const int Jn = Jc + 32;
            const unsigned short* g0 = qb + (size_t)(Jn + 2 * jp) * Dc + seg * 8;
            const unsigned short* g1 = qb + (size_t)(Jn + 16 + 2 * jp) * Dc + seg * 8;
            n00 = *(const uint4*)(g0);
            n01 = *(const uint4*)(g0 + Dc);
            n10 = *(const uint4*)(g1);
            n11 = *(const uint4*)(g1 + Dc);
        }

        // ---- scores: two 16x16 tiles, K=128 in 4 WMMA steps each ---------
        v8f s0, s1;
        #pragma unroll
        for (int e = 0; e < 8; ++e) { s0[e] = 0.0f; s1[e] = 0.0f; }
        #pragma unroll
        for (int ks = 0; ks < 4; ++ks) {
            FragBF b0, b1;   // B lane ln = column n: K chunks {8h..8h+7, 16+8h..} of row n
            const int off = ks * 32 + 8 * hh;
            b0.u[0] = *(const uint4*)&qjc[ln * QJ_STRIDE + off];
            b0.u[1] = *(const uint4*)&qjc[ln * QJ_STRIDE + off + 16];
            b1.u[0] = *(const uint4*)&qjc[(16 + ln) * QJ_STRIDE + off];
            b1.u[1] = *(const uint4*)&qjc[(16 + ln) * QJ_STRIDE + off + 16];
            s0 = __builtin_amdgcn_wmma_f32_16x16x32_bf16(
                     false, aQ[ks].v, false, b0.v, (short)0, s0, false, false);
            s1 = __builtin_amdgcn_wmma_f32_16x16x32_bf16(
                     false, aQ[ks].v, false, b1.v, (short)0, s1, false, false);
        }

        // ---- leaky + edge mask + online softmax (max only; sum via WMMA) -
        float csc[8];
        #pragma unroll
        for (int r = 0; r < 8; ++r) {
            // slope 0.2 > 0  =>  leaky_relu(x) == max(x, 0.2*x)
            float x0 = fmaxf(s0[r], LEAKY * s0[r]);
            float x1 = fmaxf(s1[r], LEAKY * s1[r]);
            const int a0 = __builtin_nontemporal_load(&adj[(size_t)adjbase[r] + Jc + ln]);
            const int a1 = __builtin_nontemporal_load(&adj[(size_t)adjbase[r] + Jc + 16 + ln]);
            x0 = (a0 >= 1 && a0 <= 4) ? x0 : MASKF;
            x1 = (a1 >= 1 && a1 <= 4) ? x1 : MASKF;

            float m = fmaxf(x0, x1);            // combine tiles, then butterfly
            DPP_MAXX(m, 1); DPP_MAXX(m, 2); DPP_MAXX(m, 4); DPP_MAXX(m, 8);
            const float mn   = fmaxf(mrun[r], m);
            const float corr = __expf(mrun[r] - mn);
            mrun[r] = mn;
            csc[r] = corr;
            ptw[(r + 8 * hh) * PT_STRIDE + ln]      = f2bf(__expf(x0 - mn));
            ptw[(r + 8 * hh) * PT_STRIDE + 16 + ln] = f2bf(__expf(x1 - mn));
        }

        // rescale only when the running max actually moved (corr != 1);
        // mn == mrun gives corr == expf(0) == 1.0 exactly, the common case.
        bool resc = false;
        #pragma unroll
        for (int r = 0; r < 8; ++r) resc = resc || (csc[r] != 1.0f);
        if (__any(resc)) {
            #pragma unroll
            for (int nt = 0; nt < 8; ++nt)
                #pragma unroll
                for (int r = 0; r < 8; ++r) acc[nt][r] *= csc[r];
            #pragma unroll
            for (int r = 0; r < 8; ++r) acc9[r] *= csc[r];
        }

        // own-wave P write -> read ordering (DS pipe is in-order per wave;
        // the wait + clobber also stops compiler reordering)
        asm volatile("s_wait_dscnt 0x0" ::: "memory");

        // ---- out += P (16x32) . V (32x128), full K=32 WMMAs --------------
        FragBF pa;          // A: lane = row ln, K run 16h..16h+15
        pa.u[0] = *(const uint4*)&ptw[ln * PT_STRIDE + 16 * hh];
        pa.u[1] = *(const uint4*)&ptw[ln * PT_STRIDE + 16 * hh + 8];

        #pragma unroll
        for (int nt = 0; nt < 8; ++nt) {
            FragBF vb;      // B: lane = column d, K(j) run 16h..16h+15
            const unsigned short* vp = &vtc[(nt * 16 + ln) * VT_STRIDE + 16 * hh];
            vb.w[0] = *(const uint2*)(vp);
            vb.w[1] = *(const uint2*)(vp + 4);
            vb.w[2] = *(const uint2*)(vp + 8);
            vb.w[3] = *(const uint2*)(vp + 12);
            acc[nt] = __builtin_amdgcn_wmma_f32_16x16x32_bf16(
                          false, pa.v, false, vb.v, (short)0, acc[nt], false, false);
        }
        // row-sum accumulator: P . ones-column
        acc9 = __builtin_amdgcn_wmma_f32_16x16x32_bf16(
                   false, pa.v, false, ones.v, (short)0, acc9, false, false);

        // ---- write tile Jp+1 into the other buffer -----------------------
        if (hasNext) {
            const int nxt = cur ^ 1;
            unsigned* vt32 = (unsigned*)&vt[nxt][0];
            #pragma unroll
            for (int t2 = 0; t2 < 2; ++t2) {
                const int jr = t2 * 16 + 2 * jp;
                const uint4 x0 = t2 ? n10 : n00;
                const uint4 x1 = t2 ? n11 : n01;
                *(uint4*)&qj[nxt][jr * QJ_STRIDE + seg * 8]       = x0;
                *(uint4*)&qj[nxt][(jr + 1) * QJ_STRIDE + seg * 8] = x1;
                const unsigned short* a0 = (const unsigned short*)&x0;
                const unsigned short* a1 = (const unsigned short*)&x1;
                #pragma unroll
                for (int e = 0; e < 8; ++e) {
                    const int d = seg * 8 + e;
                    vt32[d * (VT_STRIDE / 2) + 8 * t2 + jp] =
                        (unsigned)a0[e] | ((unsigned)a1[e] << 16);
                }
            }
        }
    }

    // ---- epilogue: recover denominators (cols 1..15 of acc9 are zero, so
    // a butterfly-add both reduces and broadcasts), normalize, store fp32 --
    float inv[8];
    #pragma unroll
    for (int r = 0; r < 8; ++r) {
        float t = acc9[r];
        DPP_ADDX(t, 1); DPP_ADDX(t, 2); DPP_ADDX(t, 4); DPP_ADDX(t, 8);
        inv[r] = 1.0f / t;
    }
    #pragma unroll
    for (int nt = 0; nt < 8; ++nt)
        #pragma unroll
        for (int r = 0; r < 8; ++r)
            __builtin_nontemporal_store(
                acc[nt][r] * inv[r],
                &out[((size_t)b * Nc + I0 + r + 8 * hh) * Dc + nt * 16 + ln]);
}

// ---------------------------------------------------------------------------
extern "C" void kernel_launch(void* const* d_in, const int* in_sizes, int n_in,
                              void* d_out, int out_size, void* d_ws, size_t ws_size,
                              hipStream_t stream)
{
    const float* hidden = (const float*)d_in[0];
    const int*   adj    = (const int*)d_in[1];
    const int*   ntype  = (const int*)d_in[2];
    const float* WQ     = (const float*)d_in[3];
    // d_in[4]/d_in[5] (W_K, W_V) are dead code in the reference (k = v = q bug)
    float* out = (float*)d_out;

    unsigned short* qbf = (unsigned short*)d_ws;   // B*N*D bf16 = 4 MB scratch

    proj_kernel<<<Bc * Nc, Dc, 0, stream>>>(hidden, ntype, WQ, qbf);
    attn_kernel<<<Bc * (Nc / 64), 128, 0, stream>>>(qbf, adj, out);
}
